// MultiHeadSelfAttention_44272522887182
// MI455X (gfx1250) — matrix operand order repaired and checked by reading, not compiled, not run on hardware
//
#include <hip/hip_runtime.h>

// ---------- types ----------
typedef __attribute__((ext_vector_type(16))) __bf16 v16bf;
typedef __attribute__((ext_vector_type(8)))  __bf16 v8bf;
typedef __attribute__((ext_vector_type(8)))  float  v8f;
typedef __attribute__((ext_vector_type(4)))  unsigned int v4u;
typedef __attribute__((ext_vector_type(8)))  int v8i_t;
typedef __attribute__((ext_vector_type(4)))  int v4i_t;
typedef __bf16 bf16_t;

// ---------- constants ----------
#define BATCH   4
#define SEQ     2048
#define DMODEL  1024
#define HEADS   16
#define DKH     64
#define PSTR    72    // LDS row stride (bf16 elems): 144B, 16B-aligned rows, shifts banks
                      // matches TDM pad: 32 DWORDs data + 4 DWORDs pad per row

// ---------- helpers ----------
__device__ static inline bf16_t f2bf(float f) {
    union { float f; unsigned int u; } a; a.f = f;
    unsigned int u = a.u;
    unsigned int r = u + 0x7FFFu + ((u >> 16) & 1u);   // round-to-nearest-even
    union { unsigned short s; bf16_t b; } c; c.s = (unsigned short)(r >> 16);
    return c.b;
}

__device__ static inline v8f wmma_bf16(v16bf a, v16bf b, v8f c) {
    return __builtin_amdgcn_wmma_f32_16x16x32_bf16(
        /*neg_a=*/false, a, /*neg_b=*/false, b,
        /*c_mod=*/(short)0, c, /*reuse_a=*/false, /*reuse_b=*/false);
}

// A fragment (16x32 bf16): lane<16 -> K {0..15}; lane>=16 -> K {16..31}
__device__ static inline v16bf load_a_frag(const bf16_t* p, int half) {
    v8bf lo = *(const v8bf*)(p + half * 16);
    v8bf hi = *(const v8bf*)(p + half * 16 + 8);
    v16bf r;
#pragma unroll
    for (int i = 0; i < 8; ++i) { r[i] = lo[i]; r[i + 8] = hi[i]; }
    return r;
}

// B fragment (32x16 bf16): lane holds 16 contiguous K at (lane>>4)*16 (caller applies)
__device__ static inline v16bf load_b_frag(const bf16_t* p) {
    v8bf lo = *(const v8bf*)(p);
    v8bf hi = *(const v8bf*)(p + 8);
    v16bf r;
#pragma unroll
    for (int i = 0; i < 8; ++i) { r[i] = lo[i]; r[i + 8] = hi[i]; }
    return r;
}

__device__ static inline v16bf ones_frag() {
    union { unsigned short s; bf16_t b; } c; c.s = 0x3F80;   // bf16 1.0
    v16bf r;
#pragma unroll
    for (int i = 0; i < 16; ++i) r[i] = c.b;
    return r;
}

// ---------- TDM: 2D tile load Global -> LDS ----------
// Loads a 64x64 bf16 tile whose rows are `stride_elems` apart in memory into LDS
// at byte offset lds_off, with 4-DWORD padding per 32-DWORD row => LDS stride 72 elems.
__device__ static inline void tdm_load_tile64(unsigned lds_off, const bf16_t* gptr,
                                              unsigned stride_elems) {
    unsigned long long ga = (unsigned long long)(uintptr_t)gptr;
    v4u g0;
    g0[0] = 1u;                                            // count=1, user descriptor
    g0[1] = lds_off;                                       // lds_addr (bytes)
    g0[2] = (unsigned)(ga & 0xFFFFFFFFu);                  // global_addr[31:0]
    g0[3] = (unsigned)((ga >> 32) & 0x01FFFFFFu)           // global_addr[56:32]
          | (2u << 30);                                    // type=2 (image)
    v8i_t g1;
    g1[0] = (int)((1u << 16)                               // data_size = 2 bytes
                | (1u << 20)                               // pad_enable
                | (4u << 22)                               // pad_interval: 32 DWORDs
                | (3u << 25));                             // pad_amount: 4 DWORDs
    g1[1] = (int)(64u << 16);                              // tensor_dim0[15:0] = 64
    g1[2] = (int)(64u << 16);                              // tensor_dim1[15:0] = 64
    g1[3] = (int)(64u << 16);                              // tile_dim0 = 64
    g1[4] = 64;                                            // tile_dim1 = 64, tile_dim2 = 0
    g1[5] = (int)stride_elems;                             // tensor_dim0_stride[31:0]
    g1[6] = 0;                                             // stride hi / dim1_stride lo
    g1[7] = 0;
    v4i_t z4 = {0, 0, 0, 0};                               // groups 2/3: 2D tile
    v8i_t z8 = {0, 0, 0, 0, 0, 0, 0, 0};
    __builtin_amdgcn_tensor_load_to_lds(g0, g1, z4, z4, z8, 0);
}

// ---------- kernel 1: fp32 -> bf16 convert (Q input) ----------
__global__ __launch_bounds__(256) void cvt_bf16_kernel(const float* __restrict__ in,
                                                       bf16_t* __restrict__ out, int n) {
    int i = blockIdx.x * 256 + threadIdx.x;
    if (i < n) out[i] = f2bf(in[i]);
}

// ---------- kernel 2: W[k][n] fp32 -> Wt[n][k] bf16 (tiled transpose) ----------
__global__ __launch_bounds__(256) void wtrans_kernel(const float* __restrict__ W,
                                                     bf16_t* __restrict__ Wt) {
    __shared__ float tile[32][33];
    int n0 = blockIdx.x * 32, k0 = blockIdx.y * 32;
    int tx = threadIdx.x & 31, ty = threadIdx.x >> 5;   // 32 x 8
#pragma unroll
    for (int r = 0; r < 32; r += 8)
        tile[ty + r][tx] = W[(size_t)(k0 + ty + r) * DMODEL + (n0 + tx)];
    __syncthreads();
#pragma unroll
    for (int r = 0; r < 32; r += 8)
        Wt[(size_t)(n0 + ty + r) * DMODEL + (k0 + tx)] = f2bf(tile[tx][ty + r]);
}

// ---------- kernel 3: projection GEMM  C[M=8192,N=1024] = A @ W + b ----------
// block tile 128x64, 8 waves (4 M-groups x 2 N-groups), wave tile 32x32.
__global__ __launch_bounds__(256) void gemm_kernel(const bf16_t* __restrict__ A,
                                                   const bf16_t* __restrict__ Wt,
                                                   const float* __restrict__ bias,
                                                   float* __restrict__ outF,     // fp32 q_a or null
                                                   bf16_t* __restrict__ outB,    // bf16 activation
                                                   int transposedV) {            // 1: store Vt[b,h,dk,s]
    int lane = threadIdx.x & 31;
    int wave = threadIdx.x >> 5;
    int wm = wave & 3, wn = wave >> 2;
    int rowBase = blockIdx.y * 128 + wm * 32;
    int colBase = blockIdx.x * 64 + wn * 32;
    int half = lane >> 4, l16 = lane & 15;

    v8f acc[2][2] = {};

    const bf16_t* arow0 = A + (size_t)(rowBase + 0  + l16) * DMODEL;
    const bf16_t* arow1 = A + (size_t)(rowBase + 16 + l16) * DMODEL;
    const bf16_t* brow0 = Wt + (size_t)(colBase + 0  + l16) * DMODEL + half * 16;
    const bf16_t* brow1 = Wt + (size_t)(colBase + 16 + l16) * DMODEL + half * 16;

    for (int kk = 0; kk < DMODEL; kk += 32) {
        v16bf fa0 = load_a_frag(arow0 + kk, half);
        v16bf fa1 = load_a_frag(arow1 + kk, half);
        v16bf fb0 = load_b_frag(brow0 + kk);
        v16bf fb1 = load_b_frag(brow1 + kk);
        acc[0][0] = wmma_bf16(fa0, fb0, acc[0][0]);
        acc[0][1] = wmma_bf16(fa0, fb1, acc[0][1]);
        acc[1][0] = wmma_bf16(fa1, fb0, acc[1][0]);
        acc[1][1] = wmma_bf16(fa1, fb1, acc[1][1]);
    }

#pragma unroll
    for (int mt = 0; mt < 2; ++mt) {
#pragma unroll
        for (int nt = 0; nt < 2; ++nt) {
            int n = colBase + nt * 16 + l16;
            float bv = bias[n];
            int m0 = rowBase + mt * 16 + half * 8;
            if (!transposedV) {
#pragma unroll
                for (int i = 0; i < 8; ++i) {
                    int m = m0 + i;
                    float v = acc[mt][nt][i] + bv;
                    if (outF) outF[(size_t)m * DMODEL + n] = v;
                    outB[(size_t)m * DMODEL + n] = f2bf(v);
                }
            } else {
                // Vt[b][h][dk][s], contiguous along s -> one 16B store
                int b  = m0 >> 11;       // /2048
                int s0 = m0 & 2047;
                int h  = n >> 6;
                int dk = n & 63;
                v8bf pk;
#pragma unroll
                for (int i = 0; i < 8; ++i) pk[i] = f2bf(acc[mt][nt][i] + bv);
                *(v8bf*)(outB + (((size_t)b * HEADS + h) * DKH + dk) * SEQ + s0) = pk;
            }
        }
    }
}

// ---------- kernel 4: attention ----------
// grid: (S/64, B*H); block: 128 threads = 4 waves; wave w owns q rows [qt*64+w*16, +16)
// K_j / V_j tiles staged into LDS by the Tensor Data Mover (one wave issues, all consume).
__global__ __launch_bounds__(128) void attn_kernel(const bf16_t* __restrict__ Qb,
                                                   const bf16_t* __restrict__ Kb,
                                                   const bf16_t* __restrict__ Vt,
                                                   const int* __restrict__ length,
                                                   float* __restrict__ outCtx) {
    __shared__ __attribute__((aligned(16))) bf16_t KT[64 * PSTR];  // keys x dk
    __shared__ __attribute__((aligned(16))) bf16_t VTl[64 * PSTR]; // dk x keys
    __shared__ __attribute__((aligned(16))) bf16_t P[64 * PSTR];   // exp(scores)

    int lane = threadIdx.x & 31;
    int wave = threadIdx.x >> 5;
    int half = lane >> 4, l16 = lane & 15;
    int qt = blockIdx.x;
    int b  = blockIdx.y >> 4;
    int h  = blockIdx.y & 15;
    int len = length[b];

    unsigned ldsK = (unsigned)(uintptr_t)&KT[0];
    unsigned ldsV = (unsigned)(uintptr_t)&VTl[0];

    // Q fragments for this wave's 16 rows (reused across all key blocks)
    const bf16_t* qrow = Qb + ((size_t)b * SEQ + qt * 64 + wave * 16 + l16) * DMODEL + h * DKH;
    v16bf fq0 = load_a_frag(qrow + 0,  half);   // dk 0..31
    v16bf fq1 = load_a_frag(qrow + 32, half);   // dk 32..63

    v8f acc[4] = {};          // context, 4 dk-tiles of 16
    v8f rs = {};              // row sums via P @ ones
    v16bf ones = ones_frag();

    int nkb = (len + 63) >> 6;                 // key blocks with any unmasked key
    if (nkb > SEQ / 64) nkb = SEQ / 64;

    for (int j = 0; j < nkb; ++j) {
        int kb = j * 64;

        __syncthreads();   // previous iteration's LDS reads complete before overwrite
        if (wave == 0) {
            // K tile: 64 keys x 64 dk, row stride DMODEL
            tdm_load_tile64(ldsK, Kb + ((size_t)b * SEQ + kb) * DMODEL + h * DKH, DMODEL);
            // V tile: 64 dk x 64 keys, row stride SEQ
            tdm_load_tile64(ldsV, Vt + (((size_t)b * HEADS + h) * DKH) * SEQ + kb, SEQ);
            __builtin_amdgcn_s_wait_tensorcnt(0);
        }
        __syncthreads();   // tiles visible to all waves

        // ---- scores tile, exp, mask, stage P to LDS ----
#pragma unroll
        for (int nt = 0; nt < 4; ++nt) {
            int key = kb + nt * 16 + l16;
            const bf16_t* krow = &KT[(nt * 16 + l16) * PSTR + half * 16];
            v16bf fk0 = load_b_frag(krow);        // dk 0..31 chunk
            v16bf fk1 = load_b_frag(krow + 32);   // dk 32..63 chunk
            v8f s = {};
            s = wmma_bf16(fq0, fk0, s);
            s = wmma_bf16(fq1, fk1, s);
            float mask = (key < len) ? 1.0f : 0.0f;
#pragma unroll
            for (int i = 0; i < 8; ++i) {
                float e = __expf(s[i] * 0.125f) * mask;   // 1/sqrt(64) = 0.125
                int pr = wave * 16 + half * 8 + i;
                P[pr * PSTR + nt * 16 + l16] = f2bf(e);
            }
        }
        // ---- context += P @ V ; rowsum += P @ ones (wave-private LDS rows) ----
#pragma unroll
        for (int c2 = 0; c2 < 2; ++c2) {
            const bf16_t* prow = &P[(wave * 16 + l16) * PSTR + c2 * 32];
            v16bf fp = load_a_frag(prow, half);
            rs = wmma_bf16(fp, ones, rs);
#pragma unroll
            for (int dt = 0; dt < 4; ++dt) {
                const bf16_t* vrow = &VTl[(dt * 16 + l16) * PSTR + c2 * 32 + half * 16];
                v16bf fv = load_b_frag(vrow);
                acc[dt] = wmma_bf16(fp, fv, acc[dt]);
            }
        }
    }

    // ---- normalize and store fp32 context ----
#pragma unroll
    for (int dt = 0; dt < 4; ++dt) {
        int d = h * DKH + dt * 16 + l16;
#pragma unroll
        for (int i = 0; i < 8; ++i) {
            int s = qt * 64 + wave * 16 + half * 8 + i;
            float inv = 1.0f / (rs[i] + 1e-8f);
            outCtx[((size_t)b * SEQ + s) * DMODEL + d] = acc[dt][i] * inv;
        }
    }
}

// ---------- host launcher ----------
extern "C" void kernel_launch(void* const* d_in, const int* in_sizes, int n_in,
                              void* d_out, int out_size, void* d_ws, size_t ws_size,
                              hipStream_t stream) {
    const float* Q      = (const float*)d_in[0];
    const int*   length = (const int*)d_in[1];
    const float* Wq = (const float*)d_in[2];
    const float* bq = (const float*)d_in[3];
    const float* Wk = (const float*)d_in[4];
    const float* bk = (const float*)d_in[5];
    const float* Wv = (const float*)d_in[6];
    const float* bv = (const float*)d_in[7];

    const size_t NELEM = (size_t)BATCH * SEQ * DMODEL;    // 8M
    float* outCtx = (float*)d_out;
    float* outQa  = (float*)d_out + NELEM;

    char* ws = (char*)d_ws;
    bf16_t* Ain = (bf16_t*)(ws);                          // 16MB bf16 input Q
    bf16_t* Qb  = (bf16_t*)(ws + ((size_t)16 << 20));     // 16MB projected q (bf16)
    bf16_t* Kb  = (bf16_t*)(ws + ((size_t)32 << 20));     // 16MB projected k (bf16)
    bf16_t* Vt  = (bf16_t*)(ws + ((size_t)48 << 20));     // 16MB projected v, [b,h,dk,s]
    bf16_t* Wtq = (bf16_t*)(ws + ((size_t)64 << 20));     // 2MB each, transposed bf16 weights
    bf16_t* Wtk = Wtq + (size_t)DMODEL * DMODEL;
    bf16_t* Wtv = Wtk + (size_t)DMODEL * DMODEL;

    cvt_bf16_kernel<<<(int)(NELEM / 256), 256, 0, stream>>>(Q, Ain, (int)NELEM);

    dim3 tg(DMODEL / 32, DMODEL / 32);
    wtrans_kernel<<<tg, 256, 0, stream>>>(Wq, Wtq);
    wtrans_kernel<<<tg, 256, 0, stream>>>(Wk, Wtk);
    wtrans_kernel<<<tg, 256, 0, stream>>>(Wv, Wtv);

    dim3 gg(DMODEL / 64, (BATCH * SEQ) / 128);
    gemm_kernel<<<gg, 256, 0, stream>>>(Ain, Wtq, bq, outQa,  Qb, 0);
    gemm_kernel<<<gg, 256, 0, stream>>>(Ain, Wtk, bk, nullptr, Kb, 0);
    gemm_kernel<<<gg, 256, 0, stream>>>(Ain, Wtv, bv, nullptr, Vt, 1);

    attn_kernel<<<dim3(SEQ / 64, BATCH * HEADS), 128, 0, stream>>>(Qb, Kb, Vt, length, outCtx);
}
